// Transformer_78941498901000
// MI455X (gfx1250) — matrix operand order repaired and checked by reading, not compiled, not run on hardware
//
#include <hip/hip_runtime.h>
#include <hip/hip_bf16.h>
#include <cstdint>

// Problem dims
#define BB 512
#define TT 128
#define CC 512
#define HH 8
#define LL 6
#define VV 13
#define NN (BB * TT)   // 65536 rows
#define DD 64

typedef _Float16 v16h __attribute__((ext_vector_type(16)));
typedef _Float16 v8h  __attribute__((ext_vector_type(8)));
typedef float    v8f  __attribute__((ext_vector_type(8)));
typedef unsigned int u32x4 __attribute__((ext_vector_type(4)));
typedef int      i32x8 __attribute__((ext_vector_type(8)));
typedef int      i32x4 __attribute__((ext_vector_type(4)));

// ---------------------------------------------------------------------------
// TDM: build a 2-D D# (ISA 08_async_tensor §8) and issue tensor_load_to_lds.
//   tensor: rows x tensor_d0 (f16, data_size=1 => 2-byte units), row stride
//   stride0 elements; tile: tile_d1 rows x tile_d0 elements, written densely
//   row-major into LDS at lds_off.
// ---------------------------------------------------------------------------
__device__ __forceinline__ void tdm_load_2d_f16(const _Float16* gptr,
                                                unsigned lds_off,
                                                int tensor_d0, int tensor_d1,
                                                int tile_d0, int tile_d1,
                                                long stride0) {
  unsigned long long ga = (unsigned long long)(uintptr_t)gptr;
  u32x4 g0;
  g0.x = 1u;                                        // count=1, is_restore=0
  g0.y = lds_off;                                   // lds_addr (bytes)
  g0.z = (unsigned)(ga & 0xFFFFFFFFu);              // global_addr[31:0]
  g0.w = (unsigned)((ga >> 32) & 0x01FFFFFFu) | (2u << 30);  // [56:32]|type=2
  i32x8 g1;
  g1[0] = (1 << 16);                                // data_size=1 (2 bytes)
  g1[1] = (int)((tensor_d0 & 0xFFFF) << 16);        // tensor_dim0[15:0]@48
  g1[2] = (int)(((unsigned)tensor_d0 >> 16) |
                (((unsigned)tensor_d1 & 0xFFFFu) << 16));  // dim0 hi | dim1 lo
  g1[3] = (int)((((unsigned)tensor_d1 >> 16) & 0xFFFFu) |
                (((unsigned)tile_d0 & 0xFFFFu) << 16));    // dim1 hi | tile0
  g1[4] = (int)((unsigned)tile_d1 & 0xFFFFu);       // tile_dim1 (tile_dim2=0)
  g1[5] = (int)((unsigned long long)stride0 & 0xFFFFFFFFull);  // stride0 lo
  g1[6] = (int)(((unsigned long long)stride0 >> 32) & 0xFFFFull);
  g1[7] = 0;                                        // tensor_dim1_stride unused
  i32x4 z4 = (i32x4){0, 0, 0, 0};                   // dims 2..4 unused
#if defined(__clang_major__) && (__clang_major__ >= 23)
  i32x8 z8 = (i32x8){0, 0, 0, 0, 0, 0, 0, 0};
  __builtin_amdgcn_tensor_load_to_lds(g0, g1, z4, z4, z8, 0);
#else
  __builtin_amdgcn_tensor_load_to_lds(g0, g1, z4, z4, 0);
#endif
}

// ---------------------------------------------------------------------------
// Embedding: x = wte[tok] + wpe[t]; write f32 + f16 copies
// ---------------------------------------------------------------------------
__global__ __launch_bounds__(256) void embed_kernel(
    const int* __restrict__ tok, const float* __restrict__ wte,
    const float* __restrict__ wpe, float* __restrict__ X,
    _Float16* __restrict__ Xh) {
  size_t idx = (size_t)blockIdx.x * 256 + threadIdx.x;  // < NN*CC
  int n = (int)(idx >> 9);
  int c = (int)(idx & 511);
  int t = n & (TT - 1);
  float v = wte[(size_t)tok[n] * CC + c] + wpe[(size_t)t * CC + c];
  X[idx] = v;
  Xh[idx] = (_Float16)v;
}

// ---------------------------------------------------------------------------
// Weight conversion: Wht[(l*4+m)*C*C + n*C + k] = (f16) W_m[l][k][n]
// ---------------------------------------------------------------------------
__global__ __launch_bounds__(256) void convert_weights_kernel(
    const float* __restrict__ Wq, const float* __restrict__ Wk,
    const float* __restrict__ Wv, const float* __restrict__ Wo,
    _Float16* __restrict__ Wht) {
  size_t idx = (size_t)blockIdx.x * 256 + threadIdx.x;  // < LL*4*CC*CC
  int l = (int)(idx / (4 * CC * CC));
  int rem = (int)(idx - (size_t)l * 4 * CC * CC);
  int m = rem / (CC * CC);
  int w = rem - m * (CC * CC);
  int n = w >> 9;
  int k = w & 511;
  const float* src = (m == 0) ? Wq : (m == 1) ? Wk : (m == 2) ? Wv : Wo;
  Wht[idx] = (_Float16)src[(size_t)l * CC * CC + (size_t)k * CC + n];
}

// ---------------------------------------------------------------------------
// GEMM: out[N,512] = A[N,512] @ W + bias, f16 in / f32 accumulate (WMMA).
// Block tile 64(M) x 128(N); 8 waves: 4(M) x 2(N); each wave 16x64 (4 WMMAs).
// Tiles staged via TDM (tensor_load_to_lds), double-buffered LDS, overlapped
// with WMMA compute; wave 0 owns the DMA + TENSORcnt, barrier publishes.
// ---------------------------------------------------------------------------
template <int STORE_HALF>
__global__ __launch_bounds__(256) void gemm_wmma_kernel(
    const _Float16* __restrict__ A, const _Float16* __restrict__ Wt,
    const float* __restrict__ bias, void* __restrict__ outp) {
  __shared__ _Float16 sX[2][64 * 32];   // A tiles: 64 rows x 32 K
  __shared__ _Float16 sW[2][128 * 32];  // W tiles: 128 cols x 32 K

  const int t = threadIdx.x;
  const int lane = t & 31;
  const int wave = t >> 5;
  const int wm = wave & 3;   // M sub-tile (0..3)
  const int wn = wave >> 2;  // N sub-tile (0..1)
  const int m0 = blockIdx.y * 64;
  const int n0 = blockIdx.x * 128;
  const int lm = lane & 15;
  const int hiq = lane >> 4;

  v8f acc[4];
#pragma unroll
  for (int f = 0; f < 4; ++f) acc[f] = (v8f){};

  const int rA = wm * 16 + lm;
  const int kb = hiq * 16;   // A frag K base within tile
  const int kb2 = hiq * 16;  // B frag K base within tile

  const unsigned offX[2] = {(unsigned)(uintptr_t)&sX[0][0],
                            (unsigned)(uintptr_t)&sX[1][0]};
  const unsigned offW[2] = {(unsigned)(uintptr_t)&sW[0][0],
                            (unsigned)(uintptr_t)&sW[1][0]};

  const int nsteps = CC / 32;  // 16

  // Preload tile 0 into buffer 0 (DMA owned by wave 0).
  if (wave == 0) {
    tdm_load_2d_f16(A + (size_t)m0 * CC, offX[0], CC, NN, 32, 64, CC);
    tdm_load_2d_f16(Wt + (size_t)n0 * CC, offW[0], CC, CC, 32, 128, CC);
    __builtin_amdgcn_s_wait_tensorcnt(0);
  }
  __syncthreads();

  for (int kk = 0; kk < nsteps; ++kk) {
    const int cur = kk & 1;
    // Kick DMA for next K-slice into the other buffer (overlaps with WMMA).
    if (wave == 0 && (kk + 1) < nsteps) {
      const int knext = (kk + 1) * 32;
      tdm_load_2d_f16(A + (size_t)m0 * CC + knext, offX[cur ^ 1], CC, NN, 32,
                      64, CC);
      tdm_load_2d_f16(Wt + (size_t)n0 * CC + knext, offW[cur ^ 1], CC, CC, 32,
                      128, CC);
    }

    // A fragment (16x32): halves 0..15 = K kb..kb+15
    v8h a_lo = *(const v8h*)&sX[cur][rA * 32 + kb];
    v8h a_hi = *(const v8h*)&sX[cur][rA * 32 + kb + 8];
    v16h afrag = __builtin_shufflevector(a_lo, a_hi, 0, 1, 2, 3, 4, 5, 6, 7, 8,
                                         9, 10, 11, 12, 13, 14, 15);
#pragma unroll
    for (int f = 0; f < 4; ++f) {
      const int rB = wn * 64 + f * 16 + lm;
      v8h b_lo = *(const v8h*)&sW[cur][rB * 32 + kb2];
      v8h b_hi = *(const v8h*)&sW[cur][rB * 32 + kb2 + 8];
      v16h bfrag = __builtin_shufflevector(b_lo, b_hi, 0, 1, 2, 3, 4, 5, 6, 7,
                                           8, 9, 10, 11, 12, 13, 14, 15);
      acc[f] = __builtin_amdgcn_wmma_f32_16x16x32_f16(
          false, afrag, false, bfrag, (short)0, acc[f], false, false);
    }

    // Publish the freshly DMA'd buffer for the next iteration.
    if (wave == 0) __builtin_amdgcn_s_wait_tensorcnt(0);
    __syncthreads();
  }

  // Epilogue: D element (m = r + 8*hiq, n = lm) per VGPR r
#pragma unroll
  for (int f = 0; f < 4; ++f) {
    const int gcol = n0 + wn * 64 + f * 16 + lm;
    const float bval = bias[gcol];
#pragma unroll
    for (int r = 0; r < 8; ++r) {
      const int grow = m0 + wm * 16 + hiq * 8 + r;
      const float val = acc[f][r] + bval;
      if (STORE_HALF) {
        ((_Float16*)outp)[(size_t)grow * CC + gcol] = (_Float16)val;
      } else {
        ((float*)outp)[(size_t)grow * CC + gcol] = val;
      }
    }
  }
}

// ---------------------------------------------------------------------------
// Head-axis attention at each (b,t): att[h,g] = q[h]·k[g]/8, softmax over g,
// hid[h] = sum_g att[h,g] v[g]. One wave per position.
// ---------------------------------------------------------------------------
__global__ __launch_bounds__(256) void attn_kernel(
    const _Float16* __restrict__ Qh, const _Float16* __restrict__ Kh,
    const _Float16* __restrict__ Vh, _Float16* __restrict__ Hh) {
  __shared__ float sq[8][CC];
  __shared__ float sk[8][CC];
  __shared__ float sv[8][CC];
  __shared__ float satt[8][64];

  const int lane = threadIdx.x & 31;
  const int wave = threadIdx.x >> 5;
  const size_t p = (size_t)blockIdx.x * 8 + wave;
  const size_t base = p * CC;

  for (int i = lane; i < CC; i += 32) {
    sq[wave][i] = (float)Qh[base + i];
    sk[wave][i] = (float)Kh[base + i];
    sv[wave][i] = (float)Vh[base + i];
  }
  __syncthreads();

  for (int pp = lane; pp < 64; pp += 32) {
    const int h = pp >> 3, g = pp & 7;
    float s = 0.0f;
#pragma unroll
    for (int d = 0; d < DD; ++d) s += sq[wave][h * DD + d] * sk[wave][g * DD + d];
    satt[wave][pp] = s * 0.125f;  // 1/sqrt(64)
  }
  __syncthreads();

  if (lane < HH) {
    const int h = lane;
    float mx = satt[wave][h * 8];
#pragma unroll
    for (int g = 1; g < 8; ++g) mx = fmaxf(mx, satt[wave][h * 8 + g]);
    float sum = 0.0f;
#pragma unroll
    for (int g = 0; g < 8; ++g) {
      float e = __expf(satt[wave][h * 8 + g] - mx);
      satt[wave][h * 8 + g] = e;
      sum += e;
    }
    const float inv = 1.0f / sum;
#pragma unroll
    for (int g = 0; g < 8; ++g) satt[wave][h * 8 + g] *= inv;
  }
  __syncthreads();

  for (int j = lane; j < CC; j += 32) {
    const int h = j >> 6, d = j & 63;
    float s = 0.0f;
#pragma unroll
    for (int g = 0; g < 8; ++g) s += satt[wave][h * 8 + g] * sv[wave][g * DD + d];
    Hh[base + j] = (_Float16)s;
  }
}

// ---------------------------------------------------------------------------
// Residual + LayerNorm: x = LN(O + x)*g + b. One wave per row; wave32 shuffles.
// ---------------------------------------------------------------------------
__global__ __launch_bounds__(256) void ln_kernel(
    const float* __restrict__ O, float* __restrict__ X,
    _Float16* __restrict__ Xh, const float* __restrict__ g,
    const float* __restrict__ bta) {
  const int lane = threadIdx.x & 31;
  const int wave = threadIdx.x >> 5;
  const size_t row = (size_t)blockIdx.x * 8 + wave;
  const size_t base = row * CC;

  float y[16];
  float s = 0.0f;
#pragma unroll
  for (int j = 0; j < 16; ++j) {
    const int i = j * 32 + lane;
    y[j] = O[base + i] + X[base + i];
    s += y[j];
  }
#pragma unroll
  for (int off = 16; off > 0; off >>= 1) s += __shfl_xor(s, off);
  const float mean = s * (1.0f / CC);

  float vs = 0.0f;
#pragma unroll
  for (int j = 0; j < 16; ++j) {
    const float d = y[j] - mean;
    vs += d * d;
  }
#pragma unroll
  for (int off = 16; off > 0; off >>= 1) vs += __shfl_xor(vs, off);
  const float rstd = rsqrtf(vs * (1.0f / CC) + 1e-5f);

#pragma unroll
  for (int j = 0; j < 16; ++j) {
    const int i = j * 32 + lane;
    const float xn = (y[j] - mean) * rstd * g[i] + bta[i];
    X[base + i] = xn;
    Xh[base + i] = (_Float16)xn;
  }
}

// ---------------------------------------------------------------------------
// LM head: logits[row, 0..12] = X[row] @ W_lm + b_lm. One wave per row.
// ---------------------------------------------------------------------------
__global__ __launch_bounds__(256) void lmhead_kernel(
    const float* __restrict__ X, const float* __restrict__ Wlm,
    const float* __restrict__ blm, float* __restrict__ logits) {
  const int lane = threadIdx.x & 31;
  const int wave = threadIdx.x >> 5;
  const size_t row = (size_t)blockIdx.x * 8 + wave;
  const size_t base = row * CC;

  float xv[16];
#pragma unroll
  for (int j = 0; j < 16; ++j) xv[j] = X[base + j * 32 + lane];

  float s[VV];
#pragma unroll
  for (int v = 0; v < VV; ++v) s[v] = 0.0f;
#pragma unroll
  for (int j = 0; j < 16; ++j) {
    const int c = j * 32 + lane;
#pragma unroll
    for (int v = 0; v < VV; ++v) s[v] += xv[j] * Wlm[(size_t)c * VV + v];
  }
#pragma unroll
  for (int v = 0; v < VV; ++v) {
#pragma unroll
    for (int off = 16; off > 0; off >>= 1) s[v] += __shfl_xor(s[v], off);
  }
  if (lane == 0) {
#pragma unroll
    for (int v = 0; v < VV; ++v) logits[row * VV + v] = s[v] + blm[v];
  }
}

// ---------------------------------------------------------------------------
// Loss: mean over batch of -log_softmax(logits[b, T-1])[label[b]]
// ---------------------------------------------------------------------------
__global__ __launch_bounds__(512) void loss_kernel(
    const float* __restrict__ logits, const int* __restrict__ label,
    float* __restrict__ out_loss) {
  __shared__ float red[512];
  const int b = threadIdx.x;
  const float* lg = logits + ((size_t)b * TT + (TT - 1)) * VV;
  float mx = lg[0];
#pragma unroll
  for (int v = 1; v < VV; ++v) mx = fmaxf(mx, lg[v]);
  float sum = 0.0f;
#pragma unroll
  for (int v = 0; v < VV; ++v) sum += __expf(lg[v] - mx);
  const float lse = mx + __logf(sum);
  red[b] = -(lg[label[b]] - lse);
  __syncthreads();
  for (int s2 = 256; s2 > 0; s2 >>= 1) {
    if (b < s2) red[b] += red[b + s2];
    __syncthreads();
  }
  if (b == 0) out_loss[0] = red[0] / (float)BB;
}

// ---------------------------------------------------------------------------
// Launch
// ---------------------------------------------------------------------------
extern "C" void kernel_launch(void* const* d_in, const int* in_sizes, int n_in,
                              void* d_out, int out_size, void* d_ws,
                              size_t ws_size, hipStream_t stream) {
  (void)in_sizes; (void)n_in; (void)out_size; (void)ws_size;

  const int*   tok  = (const int*)d_in[0];
  const int*   lab  = (const int*)d_in[1];
  const float* wte  = (const float*)d_in[2];
  const float* wpe  = (const float*)d_in[3];
  const float* Wq   = (const float*)d_in[4];
  const float* bq   = (const float*)d_in[5];
  const float* Wk   = (const float*)d_in[6];
  const float* bk   = (const float*)d_in[7];
  const float* Wv   = (const float*)d_in[8];
  const float* bv   = (const float*)d_in[9];
  const float* Wo   = (const float*)d_in[10];
  const float* bo   = (const float*)d_in[11];
  const float* lng  = (const float*)d_in[12];
  const float* lnb  = (const float*)d_in[13];
  const float* Wlm  = (const float*)d_in[14];
  const float* blm  = (const float*)d_in[15];

  // Workspace layout (bytes)
  char* ws = (char*)d_ws;
  const size_t SZ_X = (size_t)NN * CC * 4;  // 134,217,728
  const size_t SZ_H = (size_t)NN * CC * 2;  // 67,108,864
  float*    X   = (float*)(ws);
  _Float16* Xh  = (_Float16*)(ws + SZ_X);                 // also holds hid f16
  _Float16* Qh  = (_Float16*)(ws + SZ_X + SZ_H);
  _Float16* Kh  = (_Float16*)(ws + SZ_X + 2 * SZ_H);
  _Float16* Vh  = (_Float16*)(ws + SZ_X + 3 * SZ_H);
  float*    O   = (float*)(ws + SZ_X + 4 * SZ_H);
  _Float16* Wht = (_Float16*)(ws + 2 * SZ_X + 4 * SZ_H);  // 12.6 MB

  // 1. convert + transpose weights to f16
  {
    const size_t total = (size_t)LL * 4 * CC * CC;  // 6,291,456
    convert_weights_kernel<<<(unsigned)(total / 256), 256, 0, stream>>>(
        Wq, Wk, Wv, Wo, Wht);
  }
  // 2. embedding
  embed_kernel<<<(unsigned)(((size_t)NN * CC) / 256), 256, 0, stream>>>(
      tok, wte, wpe, X, Xh);

  const dim3 gemm_grid(CC / 128, NN / 64);  // (4, 1024)
  // 3. transformer layers
  for (int l = 0; l < LL; ++l) {
    const _Float16* Wl = Wht + (size_t)l * 4 * CC * CC;
    gemm_wmma_kernel<1><<<gemm_grid, 256, 0, stream>>>(
        Xh, Wl + 0 * (size_t)CC * CC, bq + (size_t)l * CC, (void*)Qh);
    gemm_wmma_kernel<1><<<gemm_grid, 256, 0, stream>>>(
        Xh, Wl + 1 * (size_t)CC * CC, bk + (size_t)l * CC, (void*)Kh);
    gemm_wmma_kernel<1><<<gemm_grid, 256, 0, stream>>>(
        Xh, Wl + 2 * (size_t)CC * CC, bv + (size_t)l * CC, (void*)Vh);
    attn_kernel<<<NN / 8, 256, 0, stream>>>(Qh, Kh, Vh, Xh);  // hid -> Xh(f16)
    gemm_wmma_kernel<0><<<gemm_grid, 256, 0, stream>>>(
        Xh, Wl + 3 * (size_t)CC * CC, bo + (size_t)l * CC, (void*)O);
    ln_kernel<<<NN / 8, 256, 0, stream>>>(O, X, Xh, lng + (size_t)l * CC,
                                          lnb + (size_t)l * CC);
  }
  // 4. LM head -> logits (first NN*VV floats of d_out)
  lmhead_kernel<<<NN / 8, 256, 0, stream>>>(X, Wlm, blm, (float*)d_out);
  // 5. loss -> d_out[NN*VV]
  loss_kernel<<<1, 512, 0, stream>>>((const float*)d_out, lab,
                                     (float*)d_out + (size_t)NN * VV);
}
